// ParticleAttentionBlock_20057497272373
// MI455X (gfx1250) — matrix operand order repaired and checked by reading, not compiled, not run on hardware
//
#include <hip/hip_runtime.h>
#include <math.h>

// ---------------------------------------------------------------------------
// CDNA5 / gfx1250 bf16 WMMA implementation of the ParticleAttentionBlock.
// B=16, N=128, C=256, H=8, D=32. Bandwidth-bound on u_ij (256 MB).
// ---------------------------------------------------------------------------

typedef __attribute__((ext_vector_type(16))) __bf16         v16bf;
typedef __attribute__((ext_vector_type(8)))  float          v8f;
typedef __attribute__((ext_vector_type(16))) unsigned short v16us;
typedef __attribute__((ext_vector_type(8)))  unsigned short v8us;

union BF16Frag { v16us u; v16bf b; };

__device__ __forceinline__ unsigned short f2bf(float f) {
  unsigned int u = __float_as_uint(f);
  u += 0x7FFFu + ((u >> 16) & 1u);          // round-to-nearest-even
  return (unsigned short)(u >> 16);
}

__device__ __forceinline__ v8f wmma_bf16(v16bf a, v16bf b, v8f c) {
  // (neg_a, A, neg_b, B, c_mod, C, reuse_a, reuse_b)
  return __builtin_amdgcn_wmma_f32_16x16x32_bf16(false, a, false, b, (short)0, c,
                                                 false, false);
}

// A fragment: 16x32 bf16 tile at (row0,k0) of row-major [*, lda] bf16 array.
// lanes 0-15: M=lane, K in {0..7,16..23}; lanes 16-31: M=lane-16, K+8.
__device__ __forceinline__ v16bf load_a_frag(const unsigned short* A, int lda,
                                             int row0, int k0, int lane) {
  int r  = row0 + (lane & 15);
  int kb = k0 + ((lane >> 4) << 3);
  const unsigned short* p = A + (size_t)r * lda + kb;
  v8us lo = *(const v8us*)(p);        // K = kb .. kb+7
  v8us hi = *(const v8us*)(p + 16);   // K = kb+16 .. kb+23
  BF16Frag f;
  f.u = __builtin_shufflevector(lo, hi, 0,1,2,3,4,5,6,7,8,9,10,11,12,13,14,15);
  return f.b;
}

// B fragment: 32x16 tile; weights stored TRANSPOSED Bt[N][K] so a column of B
// (= row of Bt) is contiguous. lanes 0-15: N=lane, K in {0..7,16..23}; lanes16-31: K+8.
__device__ __forceinline__ v16bf load_b_frag(const unsigned short* Bt, int ldb,
                                             int col0, int k0, int lane) {
  int c  = col0 + (lane & 15);
  int kb = k0 + ((lane >> 4) << 3);
  const unsigned short* p = Bt + (size_t)c * ldb + kb;
  v8us lo = *(const v8us*)(p);
  v8us hi = *(const v8us*)(p + 16);
  BF16Frag f;
  f.u = __builtin_shufflevector(lo, hi, 0,1,2,3,4,5,6,7,8,9,10,11,12,13,14,15);
  return f.b;
}

__device__ __forceinline__ v8f gemm_tile(const unsigned short* A, int lda,
                                         const unsigned short* Bt, int ldb,
                                         int row0, int col0, int K, int lane) {
  v8f acc = {};
  for (int k0 = 0; k0 < K; k0 += 32) {
    v16bf a = load_a_frag(A, lda, row0, k0, lane);
    v16bf b = load_b_frag(Bt, ldb, col0, k0, lane);
    acc = wmma_bf16(a, b, acc);
  }
  return acc;
}

// ---------------------------------------------------------------------------
// Weight prep: fp32 [K][N] row-major -> bf16 transposed [N][K].
// ---------------------------------------------------------------------------
__global__ void k_transpose_bf16(const float* __restrict__ in,
                                 unsigned short* __restrict__ out, int K, int N) {
  int idx = blockIdx.x * blockDim.x + threadIdx.x;
  if (idx >= K * N) return;
  int n = idx / K, k = idx % K;
  out[idx] = f2bf(in[(size_t)k * N + n]);
}

// w_pair [256][8] fp32 -> padded transposed bf16 [16][256] (cols 8..15 = 0).
__global__ void k_wpair_pad(const float* __restrict__ w,
                            unsigned short* __restrict__ out) {
  int idx = blockIdx.x * blockDim.x + threadIdx.x;   // 16*256
  if (idx >= 16 * 256) return;
  int h = idx / 256, k = idx % 256;
  float v = (h < 8) ? w[k * 8 + h] : 0.0f;
  out[idx] = f2bf(v);
}

// ---------------------------------------------------------------------------
// LayerNorm over C=256, one row per 256-thread block; bf16 output for WMMA.
// ---------------------------------------------------------------------------
__global__ void k_layernorm(const float* __restrict__ x,
                            const float* __restrict__ g,
                            const float* __restrict__ bta,
                            unsigned short* __restrict__ out) {
  __shared__ float red[256];
  int row = blockIdx.x, t = threadIdx.x;
  float v = x[(size_t)row * 256 + t];
  red[t] = v; __syncthreads();
  for (int s = 128; s > 0; s >>= 1) { if (t < s) red[t] += red[t + s]; __syncthreads(); }
  float mu = red[0] * (1.0f / 256.0f);
  __syncthreads();
  float d = v - mu;
  red[t] = d * d; __syncthreads();
  for (int s = 128; s > 0; s >>= 1) { if (t < s) red[t] += red[t + s]; __syncthreads(); }
  float inv = rsqrtf(red[0] * (1.0f / 256.0f) + 1e-5f);
  out[(size_t)row * 256 + t] = f2bf(d * inv * g[t] + bta[t]);
}

// ---------------------------------------------------------------------------
// QKV GEMM: [2048,256] x [256,768]. Scatter into q(b,h,n,d) (pre-scaled by
// D^-0.5), k(b,h,n,d), and vT(b,h,d,n) bf16 buffers for the attention kernel.
// ---------------------------------------------------------------------------
__global__ void k_gemm_qkv(const unsigned short* __restrict__ A,
                           const unsigned short* __restrict__ Bt,
                           unsigned short* __restrict__ qm,
                           unsigned short* __restrict__ km,
                           unsigned short* __restrict__ vT) {
  const int Ntiles = 48;
  int wid = (blockIdx.x * blockDim.x + threadIdx.x) >> 5;
  int lane = threadIdx.x & 31;
  int row0 = (wid / Ntiles) * 16, col0 = (wid % Ntiles) * 16;
  v8f acc = gemm_tile(A, 256, Bt, 256, row0, col0, 256, lane);
  const float scale = 0.17677669529663689f;  // 32^-0.5
  int ci = lane & 15, rb = (lane >> 4) << 3;
#pragma unroll
  for (int r = 0; r < 8; ++r) {
    int R = row0 + rb + r, j = col0 + ci;
    int bb = R >> 7, n = R & 127;
    int s = j >> 8, h = (j >> 5) & 7, d = j & 31;
    int bh = bb * 8 + h;
    float v = acc[r];
    if (s == 0)      qm[((size_t)bh * 128 + n) * 32 + d] = f2bf(v * scale);
    else if (s == 1) km[((size_t)bh * 128 + n) * 32 + d] = f2bf(v);
    else             vT[((size_t)bh * 32 + d) * 128 + n] = f2bf(v);
  }
}

// ---------------------------------------------------------------------------
// Pair bias: stream u_ij [262144,256] fp32 once, bf16-convert in flight,
// WMMA against padded w_pairT [16][256] -> bias[b][h][n][m] fp32.
// This is the bandwidth-critical kernel (256 MB).
// ---------------------------------------------------------------------------
__global__ void k_pair_bias(const float* __restrict__ u,
                            const unsigned short* __restrict__ wpT,
                            float* __restrict__ bias) {
  int wid = (blockIdx.x * blockDim.x + threadIdx.x) >> 5;   // 16384 row tiles
  int lane = threadIdx.x & 31;
  int row0 = wid * 16;
  int r = row0 + (lane & 15);
  int kbase = (lane >> 4) << 3;
  const float* up = u + (size_t)r * 256;
  v8f acc = {};
  for (int k0 = 0; k0 < 256; k0 += 32) {
    const float* p = up + k0 + kbase;
    if (k0 + 32 < 256) __builtin_prefetch(p + 32, 0, 0);   // global_prefetch_b8
    float4 f0 = *(const float4*)(p);
    float4 f1 = *(const float4*)(p + 4);
    float4 f2 = *(const float4*)(p + 16);
    float4 f3 = *(const float4*)(p + 20);
    BF16Frag a;
    a.u[0] = f2bf(f0.x);  a.u[1] = f2bf(f0.y);  a.u[2] = f2bf(f0.z);  a.u[3] = f2bf(f0.w);
    a.u[4] = f2bf(f1.x);  a.u[5] = f2bf(f1.y);  a.u[6] = f2bf(f1.z);  a.u[7] = f2bf(f1.w);
    a.u[8] = f2bf(f2.x);  a.u[9] = f2bf(f2.y);  a.u[10] = f2bf(f2.z); a.u[11] = f2bf(f2.w);
    a.u[12] = f2bf(f3.x); a.u[13] = f2bf(f3.y); a.u[14] = f2bf(f3.z); a.u[15] = f2bf(f3.w);
    v16bf b = load_b_frag(wpT, 256, 0, k0, lane);
    acc = wmma_bf16(a.b, b, acc);
  }
  int h = lane & 15, rb = (lane >> 4) << 3;
  if (h < 8) {
#pragma unroll
    for (int rr = 0; rr < 8; ++rr) {
      int R = row0 + rb + rr;
      int bb = R >> 14, n = (R >> 7) & 127, m = R & 127;
      bias[(((size_t)(bb * 8 + h) * 128 + n) * 128) + m] = acc[rr];
    }
  }
}

// ---------------------------------------------------------------------------
// Fused attention per (b,h): scores = Q.K^T (K=32, one WMMA chunk) + bias,
// wave-local softmax in LDS, then O = P.V (4 K-chunks) -> obuf bf16 [2048,256].
// Each of 8 waves owns a 16-row strip; no inter-wave dependency.
// ---------------------------------------------------------------------------
__global__ void k_attention(const unsigned short* __restrict__ qm,
                            const unsigned short* __restrict__ km,
                            const unsigned short* __restrict__ vT,
                            const float* __restrict__ bias,
                            const float* __restrict__ b_pair,
                            unsigned short* __restrict__ obuf) {
  extern __shared__ char smem[];
  float*          s_sc = (float*)smem;                       // 8*16*128 f32 = 64 KB
  unsigned short* s_p  = (unsigned short*)(smem + 65536);    // 8*16*128 bf16 = 32 KB
  int bh = blockIdx.x;
  int bb = bh >> 3, h = bh & 7;
  int w = threadIdx.x >> 5, lane = threadIdx.x & 31;
  int n0 = w * 16;
  const unsigned short* Q  = qm + (size_t)bh * 128 * 32;
  const unsigned short* Km = km + (size_t)bh * 128 * 32;
  const unsigned short* Vt = vT + (size_t)bh * 32 * 128;
  const float*          Bb = bias + (size_t)bh * 128 * 128;
  float bp = b_pair[h];
  int ci = lane & 15, rb = (lane >> 4) << 3;

  v16bf aq = load_a_frag(Q, 32, n0, 0, lane);                // reused for 8 tiles
  float* strip = s_sc + (size_t)w * 16 * 128;
#pragma unroll
  for (int t = 0; t < 8; ++t) {
    v16bf bk = load_b_frag(Km, 32, t * 16, 0, lane);
    v8f acc = {};
    acc = wmma_bf16(aq, bk, acc);
    int m = t * 16 + ci;
#pragma unroll
    for (int rr = 0; rr < 8; ++rr) {
      int n = n0 + rb + rr;
      strip[(rb + rr) * 128 + m] = acc[rr] + Bb[(size_t)n * 128 + m] + bp;
    }
  }
  // wave-local softmax: lanes 0..15 each own one row of the strip
  unsigned short* pst = s_p + (size_t)w * 16 * 128;
  if (lane < 16) {
    float* row = strip + lane * 128;
    float mx = -3.0e38f;
    for (int m = 0; m < 128; ++m) mx = fmaxf(mx, row[m]);
    float sum = 0.0f;
    for (int m = 0; m < 128; ++m) { float e = __expf(row[m] - mx); row[m] = e; sum += e; }
    float inv = 1.0f / sum;
    unsigned short* pr = pst + lane * 128;
    for (int m = 0; m < 128; ++m) pr[m] = f2bf(row[m] * inv);
  }
  // O = P @ V  (M=16, N=32, K=128)
#pragma unroll
  for (int dt = 0; dt < 2; ++dt) {
    v8f acc = {};
#pragma unroll
    for (int k0 = 0; k0 < 128; k0 += 32) {
      v16bf ap = load_a_frag(pst, 128, 0, k0, lane);
      v16bf bv = load_b_frag(Vt, 128, dt * 16, k0, lane);
      acc = wmma_bf16(ap, bv, acc);
    }
#pragma unroll
    for (int rr = 0; rr < 8; ++rr) {
      int n = n0 + rb + rr;
      int d = dt * 16 + ci;
      obuf[((size_t)(bb * 128 + n)) * 256 + h * 32 + d] = f2bf(acc[rr]);
    }
  }
}

// ---------------------------------------------------------------------------
// proj GEMM + bias + residual -> out1 fp32 [2048,256]
// ---------------------------------------------------------------------------
__global__ void k_gemm_proj(const unsigned short* __restrict__ A,
                            const unsigned short* __restrict__ Bt,
                            const float* __restrict__ bproj,
                            const float* __restrict__ resid,
                            float* __restrict__ out) {
  const int Ntiles = 16;
  int wid = (blockIdx.x * blockDim.x + threadIdx.x) >> 5;
  int lane = threadIdx.x & 31;
  int row0 = (wid / Ntiles) * 16, col0 = (wid % Ntiles) * 16;
  v8f acc = gemm_tile(A, 256, Bt, 256, row0, col0, 256, lane);
  int ci = lane & 15, rb = (lane >> 4) << 3;
#pragma unroll
  for (int r = 0; r < 8; ++r) {
    int R = row0 + rb + r, j = col0 + ci;
    size_t idx = (size_t)R * 256 + j;
    out[idx] = acc[r] + bproj[j] + resid[idx];
  }
}

// ---------------------------------------------------------------------------
// mlp1 GEMM + bias + exact GELU -> act bf16 [2048,1024]
// ---------------------------------------------------------------------------
__global__ void k_gemm_mlp1(const unsigned short* __restrict__ A,
                            const unsigned short* __restrict__ Bt,
                            const float* __restrict__ bias1,
                            unsigned short* __restrict__ act) {
  const int Ntiles = 64;
  int wid = (blockIdx.x * blockDim.x + threadIdx.x) >> 5;
  int lane = threadIdx.x & 31;
  int row0 = (wid / Ntiles) * 16, col0 = (wid % Ntiles) * 16;
  v8f acc = gemm_tile(A, 256, Bt, 256, row0, col0, 256, lane);
  int ci = lane & 15, rb = (lane >> 4) << 3;
#pragma unroll
  for (int r = 0; r < 8; ++r) {
    int R = row0 + rb + r, j = col0 + ci;
    float t = acc[r] + bias1[j];
    float g = 0.5f * t * (1.0f + erff(t * 0.70710678118654752f));
    act[(size_t)R * 1024 + j] = f2bf(g);
  }
}

// ---------------------------------------------------------------------------
// mlp2 GEMM (K=1024) + bias + residual -> d_out fp32 [2048,256]
// ---------------------------------------------------------------------------
__global__ void k_gemm_mlp2(const unsigned short* __restrict__ A,
                            const unsigned short* __restrict__ Bt,
                            const float* __restrict__ bias2,
                            const float* __restrict__ resid,
                            float* __restrict__ out) {
  const int Ntiles = 16;
  int wid = (blockIdx.x * blockDim.x + threadIdx.x) >> 5;
  int lane = threadIdx.x & 31;
  int row0 = (wid / Ntiles) * 16, col0 = (wid % Ntiles) * 16;
  v8f acc = gemm_tile(A, 1024, Bt, 1024, row0, col0, 1024, lane);
  int ci = lane & 15, rb = (lane >> 4) << 3;
#pragma unroll
  for (int r = 0; r < 8; ++r) {
    int R = row0 + rb + r, j = col0 + ci;
    size_t idx = (size_t)R * 256 + j;
    out[idx] = acc[r] + bias2[j] + resid[idx];
  }
}

// ---------------------------------------------------------------------------
extern "C" void kernel_launch(void* const* d_in, const int* in_sizes, int n_in,
                              void* d_out, int out_size, void* d_ws, size_t ws_size,
                              hipStream_t stream) {
  const float* x      = (const float*)d_in[0];
  const float* u_ij   = (const float*)d_in[1];
  const float* w_qkv  = (const float*)d_in[2];
  const float* w_proj = (const float*)d_in[3];
  const float* b_proj = (const float*)d_in[4];
  const float* g1     = (const float*)d_in[5];
  const float* b1     = (const float*)d_in[6];
  const float* g2     = (const float*)d_in[7];
  const float* b2     = (const float*)d_in[8];
  const float* w_mlp1 = (const float*)d_in[9];
  const float* b_mlp1 = (const float*)d_in[10];
  const float* w_mlp2 = (const float*)d_in[11];
  const float* b_mlp2 = (const float*)d_in[12];
  const float* w_pair = (const float*)d_in[13];
  const float* b_pair = (const float*)d_in[14];
  float* out = (float*)d_out;

  char* wsp = (char*)d_ws;
  auto take = [&](size_t bytes) -> char* {
    char* p = wsp; wsp += (bytes + 255) & ~(size_t)255; return p;
  };
  unsigned short* xn     = (unsigned short*)take((size_t)2048 * 256 * 2);
  unsigned short* wqkvT  = (unsigned short*)take((size_t)768 * 256 * 2);
  unsigned short* wprojT = (unsigned short*)take((size_t)256 * 256 * 2);
  unsigned short* wmlp1T = (unsigned short*)take((size_t)1024 * 256 * 2);
  unsigned short* wmlp2T = (unsigned short*)take((size_t)256 * 1024 * 2);
  unsigned short* wpairT = (unsigned short*)take((size_t)16 * 256 * 2);
  unsigned short* qm     = (unsigned short*)take((size_t)128 * 128 * 32 * 2);
  unsigned short* km     = (unsigned short*)take((size_t)128 * 128 * 32 * 2);
  unsigned short* vT     = (unsigned short*)take((size_t)128 * 32 * 128 * 2);
  float*          biasb  = (float*)take((size_t)128 * 128 * 128 * 4);
  unsigned short* obuf   = (unsigned short*)take((size_t)2048 * 256 * 2);
  float*          out1   = (float*)take((size_t)2048 * 256 * 4);
  unsigned short* h2     = (unsigned short*)take((size_t)2048 * 256 * 2);
  unsigned short* act    = (unsigned short*)take((size_t)2048 * 1024 * 2);

  // weight prep (tiny)
  k_transpose_bf16<<<dim3((256 * 768 + 255) / 256), 256, 0, stream>>>(w_qkv, wqkvT, 256, 768);
  k_transpose_bf16<<<dim3((256 * 256 + 255) / 256), 256, 0, stream>>>(w_proj, wprojT, 256, 256);
  k_transpose_bf16<<<dim3((256 * 1024 + 255) / 256), 256, 0, stream>>>(w_mlp1, wmlp1T, 256, 1024);
  k_transpose_bf16<<<dim3((1024 * 256 + 255) / 256), 256, 0, stream>>>(w_mlp2, wmlp2T, 1024, 256);
  k_wpair_pad<<<dim3(16), 256, 0, stream>>>(w_pair, wpairT);

  // attention branch
  k_layernorm<<<dim3(2048), 256, 0, stream>>>(x, g1, b1, xn);
  k_gemm_qkv<<<dim3(768), 256, 0, stream>>>(xn, wqkvT, qm, km, vT);
  k_pair_bias<<<dim3(2048), 256, 0, stream>>>(u_ij, wpairT, biasb);

  (void)hipFuncSetAttribute(reinterpret_cast<const void*>(k_attention),
                            hipFuncAttributeMaxDynamicSharedMemorySize, 98304);
  k_attention<<<dim3(128), 256, 98304, stream>>>(qm, km, vT, biasb, b_pair, obuf);
  k_gemm_proj<<<dim3(256), 256, 0, stream>>>(obuf, wprojT, b_proj, x, out1);

  // MLP branch
  k_layernorm<<<dim3(2048), 256, 0, stream>>>(out1, g2, b2, h2);
  k_gemm_mlp1<<<dim3(1024), 256, 0, stream>>>(h2, wmlp1T, b_mlp1, act);
  k_gemm_mlp2<<<dim3(256), 256, 0, stream>>>(act, wmlp2T, b_mlp2, out1, out);
}
